// LLaMACausalSelfAttention_59931973649007
// MI455X (gfx1250) — matrix operand order repaired and checked by reading, not compiled, not run on hardware
//
#include <hip/hip_runtime.h>
#include <cstdint>

// Problem constants (match reference)
constexpr int Bb = 2, Tt = 2048, Cc = 2048, Hh = 16, KVv = 4, Dd = 128;
constexpr int Mrows = Bb * Tt;          // 4096
constexpr int HD = Hh * Dd;             // 2048
constexpr int KVD = KVv * Dd;           // 512

typedef __attribute__((ext_vector_type(16))) _Float16 v16h;
typedef __attribute__((ext_vector_type(8)))  float    v8f;

union FragU { v16h v; unsigned int u[8]; };

// ---------------------------------------------------------------------------
// CDNA5 async global->LDS copy (ASYNCcnt path), guarded so compile never breaks.
// Builtin prototype (probe round 2): (v4i AS1*, v4i AS3*, imm offset, imm cpol).
// ---------------------------------------------------------------------------
#if __has_builtin(__builtin_amdgcn_global_load_async_to_lds_b128)
#define HAS_ASYNC_LDS 1
typedef int vi4 __attribute__((vector_size(16)));
typedef __attribute__((address_space(1))) vi4 gvi4;
typedef __attribute__((address_space(3))) vi4 lvi4;
__device__ __forceinline__ void async_copy16(const void* g, void* l) {
  __builtin_amdgcn_global_load_async_to_lds_b128(
      (gvi4*)(uintptr_t)g, (lvi4*)(unsigned)(uintptr_t)l, 0, 0);
}
#else
#define HAS_ASYNC_LDS 0
#endif

__device__ __forceinline__ void wait_async0() {
#if __has_builtin(__builtin_amdgcn_s_wait_asynccnt)
  __builtin_amdgcn_s_wait_asynccnt(0);
#else
  asm volatile("s_wait_asynccnt 0x0" ::: "memory");
#endif
}

// ---------------------------------------------------------------------------
// WMMA fragment loaders (16-bit operands, wave32; ISA 7.12.2 layouts).
// A (16x32): lane L<16 -> row L, K pairs 0..7 / +16 ; lanes 16..31 -> K 8..15 / +16.
// B (32x16): lane n<16 -> col n, K 0..7 / +16 (2/VGPR); lanes 16..31 -> K 8..15 / +16.
//            Stored "n-major" in LDS (k contiguous) so each pair is one dword.
// C/D 16x16: lanes 0..15 -> N=lane, M=vgpr; lanes 16..31 -> N=lane-16, M=8+vgpr.
// ---------------------------------------------------------------------------
__device__ __forceinline__ v16h load_frag_a(const _Float16* base, int ld) {
  const int lane = threadIdx.x & 31;
  const int row  = lane & 15;
  const int half = lane >> 4;
  FragU f;
#pragma unroll
  for (int v = 0; v < 8; ++v) {
    const int k = ((v >> 2) << 4) + (half << 3) + ((v & 3) << 1);
    f.u[v] = *reinterpret_cast<const unsigned int*>(base + row * ld + k);
  }
  return f.v;
}

__device__ __forceinline__ v16h load_frag_b(const _Float16* base, int ld) {
  const int lane = threadIdx.x & 31;
  const int n    = lane & 15;
  const int half = lane >> 4;
  FragU f;
#pragma unroll
  for (int v = 0; v < 8; ++v) {
    const int k = ((v >> 2) << 4) + (half << 3) + ((v & 3) << 1);
    f.u[v] = *reinterpret_cast<const unsigned int*>(base + n * ld + k);
  }
  return f.v;
}

__device__ __forceinline__ v8f wmma32(v16h a, v16h b, v8f c) {
  return __builtin_amdgcn_wmma_f32_16x16x32_f16(false, a, false, b, (short)0, c,
                                                false, false);
}

// ---------------------------------------------------------------------------
// fp32 -> f16 cast (plain)
// ---------------------------------------------------------------------------
__global__ void cast_f32_f16(const float* __restrict__ in,
                             _Float16* __restrict__ out, int n) {
  int i = blockIdx.x * blockDim.x + threadIdx.x;
  const int stride = gridDim.x * blockDim.x;
  for (; i < n; i += stride) out[i] = (_Float16)in[i];
}

// ---------------------------------------------------------------------------
// fp32 [K][N] -> f16 transposed [N][K], 32x32 LDS tile transpose, block (32,8).
// ---------------------------------------------------------------------------
__global__ __launch_bounds__(256) void cast_transpose(
    const float* __restrict__ in, _Float16* __restrict__ out, int K, int N) {
  __shared__ float tile[32][33];
  const int kb = blockIdx.x * 32, nb = blockIdx.y * 32;
#pragma unroll
  for (int i = threadIdx.y; i < 32; i += 8)
    tile[i][threadIdx.x] = in[(size_t)(kb + i) * N + nb + threadIdx.x];
  __syncthreads();
#pragma unroll
  for (int i = threadIdx.y; i < 32; i += 8)
    out[(size_t)(nb + i) * K + kb + threadIdx.x] = (_Float16)tile[threadIdx.x][i];
}

// ---------------------------------------------------------------------------
// GEMM: Y[M,N](f32) = A[M,K](f16,row-major) * Bt[N,K](f16,row-major,= B^T) + bias
// 64x64 tile/block, 4 waves, K staged by 32 via async global->LDS.
// Tile row stride 40 halves = 80B (16B multiple, odd*16 -> conflict-free).
// ---------------------------------------------------------------------------
constexpr int SLD = 40;

__global__ __launch_bounds__(128) void gemm_h_f32(
    const _Float16* __restrict__ A, const _Float16* __restrict__ Bt,
    const float* __restrict__ bias, float* __restrict__ Y,
    int M, int N, int K) {
  __shared__ alignas(16) _Float16 sA[64 * SLD];
  __shared__ alignas(16) _Float16 sB[64 * SLD];
  const int tid  = threadIdx.x;
  const int wave = tid >> 5;
  const int rowBase = blockIdx.y * 64;
  const int colBase = blockIdx.x * 64;

  v8f acc[4] = {};

  for (int k0 = 0; k0 < K; k0 += 32) {
#if HAS_ASYNC_LDS
#pragma unroll
    for (int i = 0; i < 2; ++i) {
      const int idx = tid + i * 128;        // 16B chunk id, 256 per tile
      const int r = idx >> 2, c = idx & 3;  // 4 chunks per 64B row-slice
      async_copy16(A  + (size_t)(rowBase + r) * K + k0 + c * 8, sA + r * SLD + c * 8);
      async_copy16(Bt + (size_t)(colBase + r) * K + k0 + c * 8, sB + r * SLD + c * 8);
    }
    wait_async0();
#else
#pragma unroll
    for (int i = 0; i < 8; ++i) {
      const int idx = tid + i * 128;        // dword id, 1024 per tile
      const int r = idx >> 4, kk = idx & 15;
      reinterpret_cast<unsigned int*>(sA)[r * (SLD / 2) + kk] =
          *reinterpret_cast<const unsigned int*>(A + (size_t)(rowBase + r) * K + k0 + kk * 2);
      reinterpret_cast<unsigned int*>(sB)[r * (SLD / 2) + kk] =
          *reinterpret_cast<const unsigned int*>(Bt + (size_t)(colBase + r) * K + k0 + kk * 2);
    }
#endif
    if (k0 + 32 < K) {
      __builtin_prefetch(A  + (size_t)(rowBase + (tid >> 1)) * K + k0 + 32 + (tid & 1) * 16, 0, 0);
      __builtin_prefetch(Bt + (size_t)(colBase + (tid >> 1)) * K + k0 + 32 + (tid & 1) * 16, 0, 0);
    }
    __syncthreads();

    const v16h a = load_frag_a(sA + (wave * 16) * SLD, SLD);
#pragma unroll
    for (int j = 0; j < 4; ++j) {
      const v16h b = load_frag_b(sB + (j * 16) * SLD, SLD);
      acc[j] = wmma32(a, b, acc[j]);
    }
    __syncthreads();
  }

  const int lane = tid & 31;
  const int n0 = lane & 15;
  const int m0 = (lane >> 4) * 8;
#pragma unroll
  for (int j = 0; j < 4; ++j) {
#pragma unroll
    for (int r = 0; r < 8; ++r) {
      const int row = rowBase + wave * 16 + m0 + r;
      const int col = colBase + j * 16 + n0;
      Y[(size_t)row * N + col] = acc[j][r] + bias[col];
    }
  }
}

// ---------------------------------------------------------------------------
// Fused RMSNorm + RoPE (interleaved pairs), one wave per (b,t,head) row of 128.
// ---------------------------------------------------------------------------
__global__ __launch_bounds__(256) void norm_rope_kernel(
    const float* __restrict__ in, const float* __restrict__ w,
    const float* __restrict__ fcos, const float* __restrict__ fsin,
    _Float16* __restrict__ out, int heads, int rows) {
  const int wid = (blockIdx.x * blockDim.x + threadIdx.x) >> 5;
  if (wid >= rows) return;
  const int lane = threadIdx.x & 31;
  const int t = (wid / heads) % Tt;
  const int d0 = lane * 4;

  const float4 g = *reinterpret_cast<const float4*>(in + (size_t)wid * Dd + d0);
  float ss = g.x * g.x + g.y * g.y + g.z * g.z + g.w * g.w;
#pragma unroll
  for (int m = 16; m >= 1; m >>= 1) ss += __shfl_xor(ss, m, 32);
  const float r = rsqrtf(ss * (1.0f / 128.0f) + 1e-5f);

  const float a0 = g.x * r * w[d0 + 0], b0 = g.y * r * w[d0 + 1];
  const float a1 = g.z * r * w[d0 + 2], b1 = g.w * r * w[d0 + 3];
  const int i0 = lane * 2, i1 = lane * 2 + 1;
  const float c0 = fcos[t * 64 + i0], s0 = fsin[t * 64 + i0];
  const float c1 = fcos[t * 64 + i1], s1 = fsin[t * 64 + i1];

  union { uint2 u; _Float16 h[4]; } pk;
  pk.h[0] = (_Float16)(a0 * c0 - b0 * s0);
  pk.h[1] = (_Float16)(a0 * s0 + b0 * c0);
  pk.h[2] = (_Float16)(a1 * c1 - b1 * s1);
  pk.h[3] = (_Float16)(a1 * s1 + b1 * c1);
  *reinterpret_cast<uint2*>(out + (size_t)wid * Dd + d0) = pk.u;
}

// ---------------------------------------------------------------------------
// Causal GQA flash attention. Block = 4 waves = 64 queries of one (b,h).
// Per 32-key step: S = Q K^T (8 WMMAs), online softmax, O += P V (8 WMMAs).
// sK [key][d] stride 136 halves (272B = 17*16B): async-stageable + conflict-free.
// sV [d][key] stride 40 halves (transposed, manual staging).
// ---------------------------------------------------------------------------
constexpr int SKLD = 136;
constexpr int SVLD = 40;
constexpr int SPLD = 40;

__global__ __launch_bounds__(128) void attn_kernel(
    const _Float16* __restrict__ qh, const _Float16* __restrict__ kh,
    const _Float16* __restrict__ vh, _Float16* __restrict__ oh) {
  __shared__ alignas(16) _Float16 sK[32 * SKLD];
  __shared__ alignas(16) _Float16 sV[128 * SVLD];
  __shared__ alignas(16) _Float16 sP[4][16 * SPLD];

  const int tid = threadIdx.x, wave = tid >> 5, lane = tid & 31;
  const int nqt = Tt / 64;
  const int qt = blockIdx.x % nqt;
  const int h  = (blockIdx.x / nqt) % Hh;
  const int b  = blockIdx.x / (nqt * Hh);
  const int kvh = h >> 2;                 // H/KV = 4
  const int qBase = qt * 64;
  const int q0 = qBase + wave * 16;

  // Q fragments straight from global (row stride H*D halves, dword aligned)
  const _Float16* qbase = qh + (((size_t)(b * Tt + q0) * Hh) + h) * Dd;
  v16h aQ[4];
#pragma unroll
  for (int kk = 0; kk < 4; ++kk) aQ[kk] = load_frag_a(qbase + kk * 32, Hh * Dd);

  v8f O[8] = {};
  float mrow[8], lrow[8];
#pragma unroll
  for (int r = 0; r < 8; ++r) { mrow[r] = -1e30f; lrow[r] = 0.0f; }

  const int n0 = lane & 15;
  const int m0 = (lane >> 4) * 8;
  const float scale = 0.08838834764831845f;   // 1/sqrt(128)
  const int nsteps = (qBase + 64) / 32;       // keys 0 .. qBase+63

  for (int st = 0; st < nsteps; ++st) {
    const int keyBase = st * 32;
    // ---- stage K tile: sK[key][d] ----
#if HAS_ASYNC_LDS
#pragma unroll
    for (int i = 0; i < 4; ++i) {
      const int idx = tid + i * 128;          // 16B chunk id, 512 total
      const int key = idx >> 4;
      const int du  = idx & 15;               // 16B units within 256B key row
      async_copy16(
          kh + (((size_t)(b * Tt + keyBase + key) * KVv) + kvh) * Dd + du * 8,
          sK + key * SKLD + du * 8);
    }
#else
#pragma unroll
    for (int i = 0; i < 16; ++i) {
      const int idx = tid + i * 128;          // dword id, 2048 total
      const int key = idx >> 6;
      const int du  = idx & 63;
      reinterpret_cast<unsigned int*>(sK)[key * (SKLD / 2) + du] =
          reinterpret_cast<const unsigned int*>(kh)[
              (((size_t)(b * Tt + keyBase + key) * KVv) + kvh) * 64 + du];
    }
#endif
    // ---- stage V transposed: sV[d][key] (dword global reads) ----
#pragma unroll
    for (int i = 0; i < 16; ++i) {
      const int idx = tid + i * 128;          // dword id, 2048 total
      const int key = idx >> 6;
      const int dd  = (idx & 63) * 2;
      union { unsigned int u; _Float16 h[2]; } cv;
      cv.u = reinterpret_cast<const unsigned int*>(vh)[
          (((size_t)(b * Tt + keyBase + key) * KVv) + kvh) * 64 + (idx & 63)];
      sV[(dd + 0) * SVLD + key] = cv.h[0];
      sV[(dd + 1) * SVLD + key] = cv.h[1];
    }
    if (st + 1 < nsteps) {
      const size_t nro =
          (((size_t)(b * Tt + keyBase + 32 + (tid >> 2)) * KVv) + kvh) * Dd +
          (tid & 3) * 32;
      __builtin_prefetch(kh + nro, 0, 0);
      __builtin_prefetch(vh + nro, 0, 0);
    }
#if HAS_ASYNC_LDS
    wait_async0();
#endif
    __syncthreads();

    // ---- S = Q * K^T, two 16-key sub-tiles ----
    v8f s[2];
#pragma unroll
    for (int nt = 0; nt < 2; ++nt) {
      v8f a = {};
#pragma unroll
      for (int kk = 0; kk < 4; ++kk) {
        const v16h bK = load_frag_b(sK + (nt * 16) * SKLD + kk * 32, SKLD);
        a = wmma32(aQ[kk], bK, a);
      }
      s[nt] = a;
    }

    // ---- scale + causal mask + online softmax ----
    float alpha[8];
#pragma unroll
    for (int r = 0; r < 8; ++r) {
      const int q = q0 + m0 + r;
      float s0v = s[0][r] * scale;
      float s1v = s[1][r] * scale;
      if (keyBase + n0 > q)      s0v = -1e30f;
      if (keyBase + 16 + n0 > q) s1v = -1e30f;
      float rm = fmaxf(s0v, s1v);
#pragma unroll
      for (int msk = 8; msk >= 1; msk >>= 1) rm = fmaxf(rm, __shfl_xor(rm, msk, 32));
      const float mnew = fmaxf(mrow[r], rm);
      alpha[r] = __expf(mrow[r] - mnew);
      const float p0 = __expf(s0v - mnew);
      const float p1 = __expf(s1v - mnew);
      s[0][r] = p0; s[1][r] = p1;
      float sum = p0 + p1;
#pragma unroll
      for (int msk = 8; msk >= 1; msk >>= 1) sum += __shfl_xor(sum, msk, 32);
      lrow[r] = lrow[r] * alpha[r] + sum;
      mrow[r] = mnew;
    }
#pragma unroll
    for (int j = 0; j < 8; ++j)
#pragma unroll
      for (int r = 0; r < 8; ++r) O[j][r] *= alpha[r];

    // ---- C-layout -> A-layout for P via per-wave LDS round trip ----
    _Float16* pw = sP[wave];
#pragma unroll
    for (int nt = 0; nt < 2; ++nt)
#pragma unroll
      for (int r = 0; r < 8; ++r)
        pw[(m0 + r) * SPLD + nt * 16 + n0] = (_Float16)s[nt][r];
    asm volatile("s_wait_dscnt 0x0" ::: "memory");   // CDNA5 split DS counter

    const v16h pf = load_frag_a(pw, SPLD);
#pragma unroll
    for (int j = 0; j < 8; ++j) {
      const v16h bV = load_frag_b(sV + (j * 16) * SVLD, SVLD);
      O[j] = wmma32(pf, bV, O[j]);
    }
    __syncthreads();
  }

  // ---- normalize and store: oh[b,t, h*128 + d] ----
#pragma unroll
  for (int r = 0; r < 8; ++r) {
    const float inv = 1.0f / lrow[r];
    const int t = q0 + m0 + r;
#pragma unroll
    for (int j = 0; j < 8; ++j)
      oh[(size_t)(b * Tt + t) * HD + h * Dd + j * 16 + n0] =
          (_Float16)(O[j][r] * inv);
  }
}

// ---------------------------------------------------------------------------
// Host launcher
// ---------------------------------------------------------------------------
extern "C" void kernel_launch(void* const* d_in, const int* in_sizes, int n_in,
                              void* d_out, int out_size, void* d_ws, size_t ws_size,
                              hipStream_t stream) {
  (void)in_sizes; (void)n_in; (void)out_size; (void)ws_size;
  const float* x  = (const float*)d_in[0];
  const float* Wq = (const float*)d_in[1];
  const float* bq = (const float*)d_in[2];
  const float* Wk = (const float*)d_in[3];
  const float* bk = (const float*)d_in[4];
  const float* Wv = (const float*)d_in[5];
  const float* bv = (const float*)d_in[6];
  const float* Wo = (const float*)d_in[7];
  const float* bo = (const float*)d_in[8];
  const float* qn = (const float*)d_in[9];
  const float* kn = (const float*)d_in[10];
  const float* fc = (const float*)d_in[11];
  const float* fs = (const float*)d_in[12];
  // d_in[13] = dense mask: unused, causality computed analytically
  float* out = (float*)d_out;

  char* ws = (char*)d_ws;
  size_t off = 0;
  auto alloc = [&](size_t bytes) -> void* {
    void* p = ws + off;
    off = (off + bytes + 255) & ~(size_t)255;
    return p;
  };

  _Float16* x_h  = (_Float16*)alloc((size_t)Mrows * Cc * 2);
  _Float16* Wq_h = (_Float16*)alloc((size_t)Cc * HD  * 2);   // transposed [HD][Cc]
  _Float16* Wk_h = (_Float16*)alloc((size_t)Cc * KVD * 2);   // transposed [KVD][Cc]
  _Float16* Wv_h = (_Float16*)alloc((size_t)Cc * KVD * 2);   // transposed
  _Float16* Wo_h = (_Float16*)alloc((size_t)HD * Cc  * 2);   // transposed [Cc][HD]
  float*    q32  = (float*)   alloc((size_t)Mrows * HD  * 4);
  float*    k32  = (float*)   alloc((size_t)Mrows * KVD * 4);
  float*    v32  = (float*)   alloc((size_t)Mrows * KVD * 4);
  _Float16* q_h  = (_Float16*)alloc((size_t)Mrows * HD  * 2);
  _Float16* k_h  = (_Float16*)alloc((size_t)Mrows * KVD * 2);
  _Float16* v_h  = (_Float16*)alloc((size_t)Mrows * KVD * 2);
  _Float16* o_h  = (_Float16*)alloc((size_t)Mrows * HD  * 2);

  // 1) fp32 -> f16 casts; weights transposed to [N][K] once
  cast_f32_f16 <<<4096, 256, 0, stream>>>(x, x_h, Mrows * Cc);
  cast_transpose<<<dim3(Cc / 32, HD  / 32), dim3(32, 8), 0, stream>>>(Wq, Wq_h, Cc, HD);
  cast_transpose<<<dim3(Cc / 32, KVD / 32), dim3(32, 8), 0, stream>>>(Wk, Wk_h, Cc, KVD);
  cast_transpose<<<dim3(Cc / 32, KVD / 32), dim3(32, 8), 0, stream>>>(Wv, Wv_h, Cc, KVD);
  cast_transpose<<<dim3(HD / 32, Cc  / 32), dim3(32, 8), 0, stream>>>(Wo, Wo_h, HD, Cc);

  // 2) QKV projections (WMMA)
  gemm_h_f32<<<dim3(HD / 64,  Mrows / 64), 128, 0, stream>>>(x_h, Wq_h, bq, q32, Mrows, HD,  Cc);
  gemm_h_f32<<<dim3(KVD / 64, Mrows / 64), 128, 0, stream>>>(x_h, Wk_h, bk, k32, Mrows, KVD, Cc);
  gemm_h_f32<<<dim3(KVD / 64, Mrows / 64), 128, 0, stream>>>(x_h, Wv_h, bv, v32, Mrows, KVD, Cc);

  // 3) RMSNorm + RoPE on q,k ; plain cast for v
  const int qrows = Mrows * Hh;    // 65536
  const int krows = Mrows * KVv;   // 16384
  norm_rope_kernel<<<(qrows * 32) / 256, 256, 0, stream>>>(q32, qn, fc, fs, q_h, Hh,  qrows);
  norm_rope_kernel<<<(krows * 32) / 256, 256, 0, stream>>>(k32, kn, fc, fs, k_h, KVv, krows);
  cast_f32_f16<<<2048, 256, 0, stream>>>(v32, v_h, Mrows * KVD);

  // 4) causal GQA flash attention (WMMA)
  attn_kernel<<<Bb * Hh * (Tt / 64), 128, 0, stream>>>(q_h, k_h, v_h, o_h);

  // 5) output projection (WMMA) -> fp32 d_out
  gemm_h_f32<<<dim3(Cc / 64, Mrows / 64), 128, 0, stream>>>(o_h, Wo_h, bo, out, Mrows, Cc, HD);
}
